// FluxGNN1D_39994735460755
// MI455X (gfx1250) — matrix operand order repaired and checked by reading, not compiled, not run on hardware
//
#include <hip/hip_runtime.h>
#include <math.h>

// ---------------------------------------------------------------------------
// FluxGNN1D fused solver for gfx1250 (MI455X).
//  - One kernel, z kept in LDS across all 9 sub-steps (temporal tiling, halo=9)
//  - MLP via v_wmma_f32_16x16x32_f16, f32 accumulate, f32 state/update/decode
//  - Branchless activations on native v_exp_f32 / v_rcp_f32 (no EXEC churn)
//  - grid = (N/TILE, B) = (64, 8), block = 256 threads = 8 wave32
// ---------------------------------------------------------------------------

typedef __attribute__((ext_vector_type(16))) _Float16 v16h;
typedef __attribute__((ext_vector_type(8)))  float    v8f;

#define BATCH   8
#define NCELL   8192
#define LAT     64
#define TILE    128
#define HALO    9
#define ECELL   (TILE + 2*HALO)      // 146 cells incl. halo
#define NIF     (ECELL - 1)          // 145 interfaces
#define MT      10                   // ceil(145/16) M-tiles of 16 interfaces
#define NSTEPS  4
#define REPS    3
#define XSTR    136                  // halves/row of x buffer (128 + 8 pad)
#define FXSTR   68                   // floats/row of flux buffer (aliases x)
#define HSTR    72                   // halves/row of h buffer (64 + 8 pad)

// LDS carve (all 16B aligned)
#define OFF_ZS   0                       // 146*64*4      = 37376
#define OFF_WF   37376                   // 32*32*16*2    = 32768
#define OFF_XB   70144                   // 160*136*2     = 43520 (fx aliases)
#define OFF_HB   113664                  // 160*72*2      = 23040 (us aliases)
#define OFF_WV   136704                  // 64*4          = 256
#define SMEM_BYTES 136960

__device__ __forceinline__ v8f wmma_f16(v16h a, v16h b, v8f c) {
    return __builtin_amdgcn_wmma_f32_16x16x32_f16(
        /*neg_a=*/false, a, /*neg_b=*/false, b,
        /*c_mod=*/(short)0, c, /*reuse_a=*/false, /*reuse_b=*/false);
}

#define LOG2E 1.4426950408889634f

// branchless tanh on v_exp_f32 / v_rcp_f32
__device__ __forceinline__ float tanh_fast(float x) {
    float xc = fminf(fmaxf(x, -9.01f), 9.01f);          // e^{2x} stays finite
    float e  = __builtin_amdgcn_exp2f(xc * (2.0f * LOG2E));
    return (e - 1.0f) * __builtin_amdgcn_rcpf(e + 1.0f);
}

// branchless erf, Abramowitz–Stegun 7.1.26 (|err| < 1.5e-7)
__device__ __forceinline__ float erf_fast(float x) {
    float a = fabsf(x);
    float t = __builtin_amdgcn_rcpf(fmaf(0.3275911f, a, 1.0f));
    float p = fmaf(t, 1.061405429f, -1.453152027f);
    p = fmaf(t, p, 1.421413741f);
    p = fmaf(t, p, -0.284496736f);
    p = fmaf(t, p, 0.254829592f);
    p = p * t;
    float e = __builtin_amdgcn_exp2f(-(a * a) * LOG2E); // exp(-x^2)
    float r = fmaf(-p, e, 1.0f);
    return copysignf(r, x);
}

__device__ __forceinline__ float gelu_fast(float x) {
    return 0.5f * x * (1.0f + erf_fast(x * 0.70710678118654752440f));
}

// A operand: row-major f16 LDS tile, 16 rows, K-tile kt (32 wide).
// lane layout: m = lane&15, koff = (lane>>4)*8 ; halves [koff..koff+7] and
// [koff+16..koff+23] of that row (ISA 16-bit A 16x32 layout).
__device__ __forceinline__ v16h load_afrag(const _Float16* rowbase, int strideH,
                                           int kt, int lane) {
    const _Float16* p = rowbase + (lane & 15) * strideH + kt * 32 + ((lane >> 4) << 3);
    union { v16h v; uint4 u[2]; } r;
    r.u[0] = *(const uint4*)(p);
    r.u[1] = *(const uint4*)(p + 16);
    return r.v;
}

// B operand: pre-swizzled fragment f, 16 contiguous halves per lane.
__device__ __forceinline__ v16h load_bfrag(const _Float16* wf, int f, int lane) {
    const _Float16* p = wf + (((f << 5) + lane) << 4);
    union { v16h v; uint4 u[2]; } r;
    r.u[0] = *(const uint4*)(p);
    r.u[1] = *(const uint4*)(p + 8);
    return r.v;
}

__global__ __launch_bounds__(256)
void fluxgnn_kernel(const float* __restrict__ u0,
                    const float* __restrict__ wenc,
                    const float* __restrict__ W1, const float* __restrict__ b1,
                    const float* __restrict__ W2, const float* __restrict__ b2,
                    const float* __restrict__ W3, const float* __restrict__ b3,
                    const float* __restrict__ pdt, const float* __restrict__ pdx,
                    float* __restrict__ out)
{
    extern __shared__ char smem[];
    float*    zs = (float*)   (smem + OFF_ZS);   // [146][64] f32 latent state
    _Float16* wf = (_Float16*)(smem + OFF_WF);   // weight B-fragments
    _Float16* xb = (_Float16*)(smem + OFF_XB);   // [160][136] f16 features
    float*    fx = (float*)   (smem + OFF_XB);   // [160][68]  f32 flux (alias)
    _Float16* hb = (_Float16*)(smem + OFF_HB);   // [160][72]  f16 activations
    float*    us = (float*)   (smem + OFF_HB);   // u0 tile staging (alias)
    float*    wv = (float*)   (smem + OFF_WV);   // encoder weight [64]

    const int tid  = threadIdx.x;
    const int lane = tid & 31;
    const int wid  = tid >> 5;
    const int tile_start = blockIdx.x * TILE;
    const int bi = blockIdx.y;
    // edge-pad ghost semantics via clamped source index (exact vs reference)
    const int srcLo = (tile_start == 0) ? HALO : 0;
    const int srcHi = (tile_start + TILE == NCELL) ? (HALO + TILE - 1) : (ECELL - 1);

    // ---- one-time setup: swizzle weights into WMMA B-fragment layout -------
    for (int t = tid; t < 32 * 32; t += 256) {
        int f = t >> 5, ln = t & 31;
        const float* src; int kt, nt;
        if (f < 16)      { src = W1; kt = f >> 2;        nt = f & 3; }
        else if (f < 24) { src = W2; kt = (f - 16) >> 2; nt = (f - 16) & 3; }
        else             { src = W3; kt = (f - 24) >> 2; nt = (f - 24) & 3; }
        int n  = nt * 16 + (ln & 15);
        int kb = kt * 32 + ((ln >> 4) << 3);
        _Float16* dst = wf + (((f << 5) + ln) << 4);
#pragma unroll
        for (int i = 0; i < 8; ++i) {
            dst[i]     = (_Float16)src[(kb + i)      * LAT + n];
            dst[8 + i] = (_Float16)src[(kb + 16 + i) * LAT + n];
        }
    }
    for (int t = tid; t < LAT; t += 256) wv[t] = wenc[t];
    for (int t = tid; t < ECELL; t += 256) {
        int g = tile_start - HALO + t;
        g = (g < 0) ? 0 : ((g > NCELL - 1) ? NCELL - 1 : g);
        us[t] = u0[(size_t)bi * NCELL + g];
    }

    const int rbase = (lane >> 4) << 3;   // D rows: 0 or 8 within tile
    const int cloc  = lane & 15;          // D col within 16-tile

    // loop-invariant per-lane biases (straight from global, once)
    float bia1[4], bia2[4], bia3[4];
#pragma unroll
    for (int nt = 0; nt < 4; ++nt) {
        bia1[nt] = b1[nt * 16 + cloc];
        bia2[nt] = b2[nt * 16 + cloc];
        bia3[nt] = b3[nt * 16 + cloc];
    }
    __syncthreads();

    float sw = 0.f;
#pragma unroll
    for (int l = 0; l < LAT; ++l) sw += wv[l] * wv[l];
    const float invden = 1.0f / (sw + 1e-8f);
    const float lam = (pdt[0] / (float)REPS) / pdx[0];

    // encode: z = u0 * w_enc
    for (int t = tid; t < ECELL * LAT; t += 256)
        zs[t] = us[t >> 6] * wv[t & 63];
    __syncthreads();

    // decode snapshot t=0
    if (tid < TILE) {
        float acc = 0.f;
#pragma unroll
        for (int k = 0; k < LAT; ++k) {
            int l = (k + tid) & 63;                 // bank-conflict-free rotation
            acc += zs[(HALO + tid) * LAT + l] * wv[l];
        }
        out[((size_t)bi * NSTEPS + 0) * NCELL + tile_start + tid] = acc * invden;
    }

    for (int st = 1; st < NSTEPS; ++st) {
        for (int rep = 0; rep < REPS; ++rep) {
            __syncthreads();   // z stable; flux buffer free for reuse
            // ---- per wave-owned M-tile: build features + 3-layer MLP ------
            for (int mt = wid; mt < MT; mt += 8) {
                // interface features [z_l+z_r | abs(z_l-z_r)] -> f16
                for (int t = lane; t < 16 * 64; t += 32) {
                    int r = t >> 6, l = t & 63;
                    int i = mt * 16 + r;
                    _Float16* xr = xb + (mt * 16 + r) * XSTR;
                    if (i < NIF) {
                        int i1 = i + 1;
                        int c0 = i  < srcLo ? srcLo : (i  > srcHi ? srcHi : i);
                        int c1 = i1 < srcLo ? srcLo : ((i1 > srcHi) ? srcHi : i1);
                        float zl = zs[c0 * LAT + l];
                        float zr = zs[c1 * LAT + l];
                        xr[l]      = (_Float16)(zl + zr);
                        xr[64 + l] = (_Float16)fabsf(zl - zr);
                    } else {
                        xr[l]      = (_Float16)0.f;
                        xr[64 + l] = (_Float16)0.f;
                    }
                }
                // layer 1: [16x128] x [128x64]
                v16h a1[4];
#pragma unroll
                for (int kt = 0; kt < 4; ++kt)
                    a1[kt] = load_afrag(xb + (mt * 16) * XSTR, XSTR, kt, lane);
#pragma unroll
                for (int nt = 0; nt < 4; ++nt) {
                    v8f acc = {0.f,0.f,0.f,0.f,0.f,0.f,0.f,0.f};
#pragma unroll
                    for (int kt = 0; kt < 4; ++kt)
                        acc = wmma_f16(a1[kt], load_bfrag(wf, kt * 4 + nt, lane), acc);
#pragma unroll
                    for (int r = 0; r < 8; ++r) {
                        float g = gelu_fast(acc[r] + bia1[nt]);
                        hb[(mt * 16 + rbase + r) * HSTR + nt * 16 + cloc] = (_Float16)g;
                    }
                }
                // layer 2: [16x64] x [64x64]  (in-place h: A loaded first)
                v16h a2[2];
#pragma unroll
                for (int kt = 0; kt < 2; ++kt)
                    a2[kt] = load_afrag(hb + (mt * 16) * HSTR, HSTR, kt, lane);
#pragma unroll
                for (int nt = 0; nt < 4; ++nt) {
                    v8f acc = {0.f,0.f,0.f,0.f,0.f,0.f,0.f,0.f};
#pragma unroll
                    for (int kt = 0; kt < 2; ++kt)
                        acc = wmma_f16(a2[kt], load_bfrag(wf, 16 + kt * 4 + nt, lane), acc);
#pragma unroll
                    for (int r = 0; r < 8; ++r) {
                        float g = gelu_fast(acc[r] + bia2[nt]);
                        hb[(mt * 16 + rbase + r) * HSTR + nt * 16 + cloc] = (_Float16)g;
                    }
                }
                // layer 3: [16x64] x [64x64] -> tanh * 0.25 -> f32 flux
                v16h a3[2];
#pragma unroll
                for (int kt = 0; kt < 2; ++kt)
                    a3[kt] = load_afrag(hb + (mt * 16) * HSTR, HSTR, kt, lane);
#pragma unroll
                for (int nt = 0; nt < 4; ++nt) {
                    v8f acc = {0.f,0.f,0.f,0.f,0.f,0.f,0.f,0.f};
#pragma unroll
                    for (int kt = 0; kt < 2; ++kt)
                        acc = wmma_f16(a3[kt], load_bfrag(wf, 24 + kt * 4 + nt, lane), acc);
#pragma unroll
                    for (int r = 0; r < 8; ++r) {
                        float f = tanh_fast(acc[r] + bia3[nt]) * 0.25f;
                        fx[(mt * 16 + rbase + r) * FXSTR + nt * 16 + cloc] = f;
                    }
                }
            }
            __syncthreads();   // all flux rows ready
            // ---- conservative update: z_i -= lam * (F_i - F_{i-1}) --------
            for (int t = tid; t < (ECELL - 2) * LAT; t += 256) {
                int i = 1 + (t >> 6), l = t & 63;
                zs[i * LAT + l] -= lam * (fx[i * FXSTR + l] - fx[(i - 1) * FXSTR + l]);
            }
        }
        __syncthreads();
        // decode snapshot st
        if (tid < TILE) {
            float acc = 0.f;
#pragma unroll
            for (int k = 0; k < LAT; ++k) {
                int l = (k + tid) & 63;
                acc += zs[(HALO + tid) * LAT + l] * wv[l];
            }
            out[((size_t)bi * NSTEPS + st) * NCELL + tile_start + tid] = acc * invden;
        }
    }
}

extern "C" void kernel_launch(void* const* d_in, const int* in_sizes, int n_in,
                              void* d_out, int out_size, void* d_ws, size_t ws_size,
                              hipStream_t stream)
{
    const float* u0   = (const float*)d_in[0];
    const float* wenc = (const float*)d_in[1];
    const float* W1   = (const float*)d_in[2];
    const float* b1   = (const float*)d_in[3];
    const float* W2   = (const float*)d_in[4];
    const float* b2   = (const float*)d_in[5];
    const float* W3   = (const float*)d_in[6];
    const float* b3   = (const float*)d_in[7];
    const float* pdt  = (const float*)d_in[8];
    const float* pdx  = (const float*)d_in[9];
    float* out = (float*)d_out;

    dim3 grid(NCELL / TILE, BATCH);
    hipLaunchKernelGGL(fluxgnn_kernel, grid, dim3(256), SMEM_BYTES, stream,
                       u0, wenc, W1, b1, W2, b2, W3, b3, pdt, pdx, out);
}
